// Attention_18133351924485
// MI455X (gfx1250) — compile-verified, matrix operand order checked by reading, not run on hardware
//
#include <hip/hip_runtime.h>
#include <hip/hip_bf16.h>

// MI455X / gfx1250, wave32. WMMA bf16 16x16x32 everywhere.
// Attention K/V chunks staged via GLOBAL_LOAD_ASYNC_TO_LDS (ASYNCcnt) when the
// toolchain exposes the builtin; sync b128+ds_store fallback otherwise.

typedef __attribute__((ext_vector_type(16))) __bf16 v16bf;
typedef __attribute__((ext_vector_type(8)))  float  v8f;
typedef int v4i_vs __attribute__((vector_size(4 * sizeof(int))));

#define B_ 4
#define L_ 2048
#define E_ 1024
#define D_ 1024
#define NEGV  (-1000000000.0f)
#define SCALE (0.03125f)      /* 1/sqrt(1024) */

#if __has_builtin(__builtin_amdgcn_global_load_async_to_lds_b128)
#define HAVE_ASYNC_LDS 1
#else
#define HAVE_ASYNC_LDS 0
#endif

union AB { v16bf v; unsigned u[8]; };

__device__ __forceinline__ unsigned short bf16b(float f) {
  union { float f; unsigned u; } x; x.f = f;
  unsigned r = x.u + 0x7fffu + ((x.u >> 16) & 1u);   // round-to-nearest-even
  return (unsigned short)(r >> 16);
}
__device__ __forceinline__ unsigned packbf2(float a, float b) {
  return (unsigned)bf16b(a) | ((unsigned)bf16b(b) << 16);
}

// ---------------------------------------------------------------------------
// Projection: out[m,n] = sum_k X[m,k]*W[n,k] + bias[n], bf16 output.
// M = B*L = 8192, N = D = 1024, K = E = 1024.
// Block: 256 threads (8 waves), 64x64 output tile, K-step 32.
// ---------------------------------------------------------------------------
__global__ __launch_bounds__(256)
void proj_bf16_wmma(const float* __restrict__ X, const float* __restrict__ W,
                    const float* __restrict__ bias, unsigned short* __restrict__ out) {
  __shared__ unsigned Xs[64 * 17];   // 64 rows x 16 bf16-pairs (+1 pad)
  __shared__ unsigned Ws[64 * 17];

  const int tid  = threadIdx.x;
  const int lane = tid & 31, wave = tid >> 5;
  const int hi = lane >> 4, lp = lane & 15;
  const int m0 = blockIdx.y * 64, n0 = blockIdx.x * 64;
  const int rloc = (wave & 3) * 16;     // wave's 16 rows within tile
  const int cloc = (wave >> 2) * 32;    // wave's 32 cols within tile

  v8f c0 = {0,0,0,0,0,0,0,0};
  v8f c1 = {0,0,0,0,0,0,0,0};

  for (int kk = 0; kk < E_; kk += 32) {
    __syncthreads();   // protect LDS reuse across K-steps
    #pragma unroll
    for (int i = 0; i < 4; ++i) {
      int idx = tid + 256 * i;          // 1024 pairs per tile
      int row = idx >> 4, pc = idx & 15;
      float2 xv = *(const float2*)(X + (size_t)(m0 + row) * E_ + kk + 2 * pc);
      Xs[row * 17 + pc] = packbf2(xv.x, xv.y);
      float2 wv = *(const float2*)(W + (size_t)(n0 + row) * E_ + kk + 2 * pc);
      Ws[row * 17 + pc] = packbf2(wv.x, wv.y);
    }
    __syncthreads();

    AB a, b0, b1;
    #pragma unroll
    for (int vv = 0; vv < 8; ++vv) {
      int pidx = (vv & 3) + 4 * hi + 8 * (vv >> 2);   // bf16-pair index within 32-K slice
      a.u[vv]  = Xs[(rloc + lp) * 17 + pidx];
      b0.u[vv] = Ws[(cloc + lp) * 17 + pidx];
      b1.u[vv] = Ws[(cloc + 16 + lp) * 17 + pidx];
    }
    c0 = __builtin_amdgcn_wmma_f32_16x16x32_bf16(false, a.v, false, b0.v, (short)0, c0, false, false);
    c1 = __builtin_amdgcn_wmma_f32_16x16x32_bf16(false, a.v, false, b1.v, (short)0, c1, false, false);
  }

  #pragma unroll
  for (int j = 0; j < 8; ++j) {
    int row  = m0 + rloc + j + 8 * hi;
    int col0 = n0 + cloc + lp;
    int col1 = col0 + 16;
    out[(size_t)row * D_ + col0] = bf16b(c0[j] + bias[col0]);
    out[(size_t)row * D_ + col1] = bf16b(c1[j] + bias[col1]);
  }
}

// ---------------------------------------------------------------------------
// Flash attention: block = (16 query rows, one batch), 256 threads (8 waves).
// Each wave owns a 128-wide slice of the 1024 output columns.
// Per 32-key chunk: K/V chunk -> LDS (async if available), QK^T via WMMA
// (contraction split across waves, ds_add_f32 reduce), online softmax with
// mask, then P@V via WMMA.
// ---------------------------------------------------------------------------
#define KROW_U 516    /* uints per LDS row: 512 data + 4 pad (16B aligned rows) */
#define KROW_H 1032

__global__ __launch_bounds__(256)
void flash_attn_wmma(const unsigned short* __restrict__ qb,
                     const unsigned short* __restrict__ kb,
                     const unsigned short* __restrict__ vb,
                     const int* __restrict__ mask,
                     float* __restrict__ out) {
  extern __shared__ __align__(16) unsigned char smem[];
  unsigned*       ks32 = (unsigned*)smem;                 // K chunk, 32 x 1024 bf16
  unsigned*       vs32 = ks32 + 32 * KROW_U;              // V chunk, 32 x 1024 bf16
  unsigned short* vs16 = (unsigned short*)vs32;
  float*          sc   = (float*)(vs32 + 32 * KROW_U);    // 16 x 32 score tile
  unsigned*       ps32 = (unsigned*)(sc + 512);           // 16 x 32 bf16 P tile
  float*          mrow = (float*)(ps32 + 256);            // running max [16]
  float*          lrow = mrow + 16;                       // running sum [16]
  float*          arow = lrow + 16;                       // rescale factors [16]

  const int tid  = threadIdx.x;
  const int lane = tid & 31, wave = tid >> 5;
  const int hi = lane >> 4, lp = lane & 15;
  const int q0 = blockIdx.x * 16;
  const int b  = blockIdx.y;

  // Per-wave q slice in registers: rows q0..q0+15, e in [wave*128, wave*128+128)
  AB qa[4];
  {
    const unsigned short* qrow = qb + ((size_t)(b * L_ + q0 + lp)) * D_ + wave * 128;
    #pragma unroll
    for (int t = 0; t < 4; ++t)
      #pragma unroll
      for (int vv = 0; vv < 8; ++vv) {
        int el = 2 * (vv & 3) + 8 * hi + 16 * (vv >> 2);
        qa[t].u[vv] = *(const unsigned*)(qrow + t * 32 + el);
      }
  }

  const v8f zero8 = {0,0,0,0,0,0,0,0};
  v8f acc[8];
  #pragma unroll
  for (int n = 0; n < 8; ++n) acc[n] = zero8;

  if (tid < 16) { mrow[tid] = -1e30f; lrow[tid] = 0.f; arow[tid] = 0.f; }

  for (int k0 = 0; k0 < L_; k0 += 32) {
    __syncthreads();   // previous chunk's consumers done (and init visible)

    // ---- cooperative K/V chunk load into LDS, zero score tile ----
    const uint4* gk = (const uint4*)(kb + ((size_t)(b * L_ + k0)) * D_);
    const uint4* gv = (const uint4*)(vb + ((size_t)(b * L_ + k0)) * D_);
#if HAVE_ASYNC_LDS
    #pragma unroll
    for (int i = 0; i < 16; ++i) {
      int idx = tid + 256 * i;          // 4096 b128 transfers per buffer
      int key = idx >> 7, cc = idx & 127;
      __builtin_amdgcn_global_load_async_to_lds_b128(
          (__attribute__((address_space(1))) v4i_vs*)(gk + (size_t)key * 128 + cc),
          (__attribute__((address_space(3))) v4i_vs*)((unsigned char*)&ks32[key * KROW_U + cc * 4]),
          0, 0);
      __builtin_amdgcn_global_load_async_to_lds_b128(
          (__attribute__((address_space(1))) v4i_vs*)(gv + (size_t)key * 128 + cc),
          (__attribute__((address_space(3))) v4i_vs*)((unsigned char*)&vs32[key * KROW_U + cc * 4]),
          0, 0);
    }
#else
    #pragma unroll
    for (int i = 0; i < 16; ++i) {
      int idx = tid + 256 * i;          // 4096 uint4 per buffer
      int key = idx >> 7, cc = idx & 127;
      uint4 kv = gk[(size_t)key * 128 + cc];
      *(uint4*)&ks32[key * KROW_U + cc * 4] = kv;
      uint4 vv = gv[(size_t)key * 128 + cc];
      *(uint4*)&vs32[key * KROW_U + cc * 4] = vv;
    }
#endif
    sc[tid] = 0.f; sc[tid + 256] = 0.f;
    if (k0 + 32 < L_) {   // prefetch next chunk (global_prefetch_b8)
      __builtin_prefetch(kb + ((size_t)(b * L_ + k0 + 32)) * D_ + tid * 128, 0, 1);
      __builtin_prefetch(vb + ((size_t)(b * L_ + k0 + 32)) * D_ + tid * 128, 0, 1);
    }
#if HAVE_ASYNC_LDS
#if __has_builtin(__builtin_amdgcn_s_wait_asynccnt)
    __builtin_amdgcn_s_wait_asynccnt(0);
#else
    asm volatile("s_wait_asynccnt 0x0" ::: "memory");
#endif
#endif
    __syncthreads();

    // ---- scores: contraction slice [wave*128, +128) of q . k^T ----
    v8f s0 = zero8, s1 = zero8;
    #pragma unroll
    for (int t = 0; t < 4; ++t) {
      AB b0, b1;
      int ebase = wave * 128 + t * 32;
      #pragma unroll
      for (int vv = 0; vv < 8; ++vv) {
        int el = 2 * (vv & 3) + 8 * hi + 16 * (vv >> 2);
        b0.u[vv] = ks32[lp * KROW_U        + ((ebase + el) >> 1)];
        b1.u[vv] = ks32[(16 + lp) * KROW_U + ((ebase + el) >> 1)];
      }
      s0 = __builtin_amdgcn_wmma_f32_16x16x32_bf16(false, qa[t].v, false, b0.v, (short)0, s0, false, false);
      s1 = __builtin_amdgcn_wmma_f32_16x16x32_bf16(false, qa[t].v, false, b1.v, (short)0, s1, false, false);
    }
    #pragma unroll
    for (int j = 0; j < 8; ++j) {        // ds_add_f32 reduce across waves
      atomicAdd(&sc[(j + 8 * hi) * 32 + lp],      s0[j]);
      atomicAdd(&sc[(j + 8 * hi) * 32 + lp + 16], s1[j]);
    }
    __syncthreads();

    // ---- online softmax update (one thread per query row) ----
    if (tid < 16) {
      int r = tid;
      float mprev = mrow[r];
      float mmax  = mprev;
      const int* mp = mask + ((size_t)(b * L_) + q0 + r) * L_ + k0;
      float svals[32];
      #pragma unroll
      for (int j = 0; j < 32; ++j) {
        float s = sc[r * 32 + j] * SCALE;
        if (mp[j] == 0) s = NEGV;
        svals[j] = s;
        mmax = fmaxf(mmax, s);
      }
      float alpha = __expf(mprev - mmax);
      float sum = 0.f;
      #pragma unroll
      for (int j = 0; j < 16; ++j) {
        float p0 = __expf(svals[2 * j]     - mmax);
        float p1 = __expf(svals[2 * j + 1] - mmax);
        sum += p0 + p1;
        ps32[r * 16 + j] = packbf2(p0, p1);
      }
      lrow[r] = lrow[r] * alpha + sum;
      mrow[r] = mmax;
      arow[r] = alpha;
    }
    __syncthreads();

    // ---- rescale accumulators, then P @ V on this wave's 128 columns ----
    AB pa;
    #pragma unroll
    for (int vv = 0; vv < 8; ++vv) {
      int pidx = (vv & 3) + 4 * hi + 8 * (vv >> 2);
      pa.u[vv] = ps32[lp * 16 + pidx];
    }
    #pragma unroll
    for (int j = 0; j < 8; ++j) {
      float a = arow[j + 8 * hi];
      #pragma unroll
      for (int n = 0; n < 8; ++n) acc[n][j] *= a;
    }
    #pragma unroll
    for (int n = 0; n < 8; ++n) {
      AB bv;
      int col = wave * 128 + n * 16 + lp;
      #pragma unroll
      for (int vv = 0; vv < 8; ++vv) {
        int kl = 2 * (vv & 3) + 8 * hi + 16 * (vv >> 2);
        unsigned lo = vs16[kl * KROW_H + col];
        unsigned hh = vs16[(kl + 1) * KROW_H + col];
        bv.u[vv] = lo | (hh << 16);
      }
      acc[n] = __builtin_amdgcn_wmma_f32_16x16x32_bf16(false, pa.v, false, bv.v, (short)0, acc[n], false, false);
    }
  }

  // ---- epilogue: divide by softmax denominator, store f32 ----
  #pragma unroll
  for (int j = 0; j < 8; ++j) {
    float linv = 1.0f / lrow[j + 8 * hi];
    int row = q0 + j + 8 * hi;
    #pragma unroll
    for (int n = 0; n < 8; ++n) {
      out[((size_t)(b * L_) + row) * D_ + wave * 128 + n * 16 + lp] = acc[n][j] * linv;
    }
  }
}

// ---------------------------------------------------------------------------
extern "C" void kernel_launch(void* const* d_in, const int* in_sizes, int n_in,
                              void* d_out, int out_size, void* d_ws, size_t ws_size,
                              hipStream_t stream) {
  const float* Q    = (const float*)d_in[0];
  const float* K    = (const float*)d_in[1];
  const float* V    = (const float*)d_in[2];
  const int*   mask = (const int*)  d_in[3];
  const float* Wq_w = (const float*)d_in[4];
  const float* Wq_b = (const float*)d_in[5];
  const float* Wk_w = (const float*)d_in[6];
  const float* Wk_b = (const float*)d_in[7];
  const float* Wv_w = (const float*)d_in[8];
  const float* Wv_b = (const float*)d_in[9];

  const size_t per = (size_t)B_ * L_ * D_;   // 8M bf16 elements per tensor
  unsigned short* qb = (unsigned short*)d_ws;            // 16 MB
  unsigned short* kb = qb + per;                          // 16 MB
  unsigned short* vb = kb + per;                          // 16 MB (48 MB total)

  dim3 pgrid(D_ / 64, (B_ * L_) / 64);   // 16 x 128 blocks
  proj_bf16_wmma<<<pgrid, 256, 0, stream>>>(Q, Wq_w, Wq_b, qb);
  proj_bf16_wmma<<<pgrid, 256, 0, stream>>>(K, Wk_w, Wk_b, kb);
  proj_bf16_wmma<<<pgrid, 256, 0, stream>>>(V, Wv_w, Wv_b, vb);

  size_t smem = (size_t)(2 * 32 * KROW_U) * 4   // K + V chunks
              + 512 * 4                          // score tile
              + 256 * 4                          // P tile (bf16)
              + 48 * 4;                          // m / l / alpha
  dim3 agrid(L_ / 16, B_);                       // 128 x 4 blocks
  flash_attn_wmma<<<agrid, 256, smem, stream>>>(qb, kb, vb, mask, (float*)d_out);
}
